// VectorQuantizer_28759101014148
// MI455X (gfx1250) — matrix operand order repaired and checked by reading, not compiled, not run on hardware
//
#include <hip/hip_runtime.h>
#include <hip/hip_bf16.h>
#include <stdint.h>

// Problem dims (fixed by reference)
#define B_ 16
#define D_ 512
#define T_ 2048
#define Q_ 8
#define N_ 1024
#define M_ (B_ * T_)   // 32768 rows

#define MTILE 64       // rows per workgroup (4 WMMA m-subtiles)

typedef __attribute__((ext_vector_type(16))) __bf16          v16bf;
typedef __attribute__((ext_vector_type(8)))  float           v8f;
typedef __attribute__((ext_vector_type(8)))  unsigned short  u16x8;

union ABFrag { u16x8 h[2]; v16bf v; };

__device__ __forceinline__ unsigned short f32_to_bf16(float f) {
    unsigned int u = __float_as_uint(f);
    unsigned int r = u + 0x7FFFu + ((u >> 16) & 1u);  // round-to-nearest-even
    return (unsigned short)(r >> 16);
}

// ---------------------------------------------------------------------------
// Prep 1: residual[b][t][d] = z[b][d][t]  (LDS-tiled transpose, 32x32 tiles)
// ---------------------------------------------------------------------------
__global__ __launch_bounds__(256) void rvq_transpose_in(
    const float* __restrict__ z, float* __restrict__ residual)
{
    __shared__ float s[32][33];
    const int b  = blockIdx.z;
    const int t0 = blockIdx.x * 32;
    const int d0 = blockIdx.y * 32;
    const int tx = threadIdx.x & 31;
    const int ty = threadIdx.x >> 5;  // 0..7
#pragma unroll
    for (int i = 0; i < 4; ++i) {
        int dl = ty + 8 * i;
        s[dl][tx] = z[((size_t)b * D_ + (d0 + dl)) * T_ + t0 + tx];  // s[d_local][t_local]
    }
    __syncthreads();
#pragma unroll
    for (int i = 0; i < 4; ++i) {
        int tl = ty + 8 * i;
        residual[((size_t)b * T_ + (t0 + tl)) * D_ + d0 + tx] = s[tx][tl];
    }
}

// ---------------------------------------------------------------------------
// Prep 2: codebook f32 -> bf16, per-row squared norms c2, zero loss partials
// ---------------------------------------------------------------------------
__global__ __launch_bounds__(256) void rvq_prep_cb(
    const float* __restrict__ cbk, unsigned short* __restrict__ cb16,
    float* __restrict__ c2, float* __restrict__ lossp)
{
    const int row  = blockIdx.x * 8 + (threadIdx.x >> 5);  // 0..Q_*N_-1
    const int lane = threadIdx.x & 31;
    const float* src = cbk + (size_t)row * D_;
    unsigned short* dst = cb16 + (size_t)row * D_;
    float s = 0.f;
#pragma unroll
    for (int i = 0; i < 16; ++i) {
        int d = i * 32 + lane;     // coalesced
        float v = src[d];
        s += v * v;
        dst[d] = f32_to_bf16(v);
    }
    for (int m = 16; m >= 1; m >>= 1) s += __shfl_xor(s, m, 32);
    if (lane == 0) c2[row] = s;
    if (blockIdx.x == 0 && threadIdx.x < Q_) lossp[threadIdx.x] = 0.f;
}

// ---------------------------------------------------------------------------
// Main per-stage kernel: fused GEMM (bf16 WMMA) + argmin + gather +
// residual update + commit-loss partial sum.
// Workgroup = 64 M-rows x all 1024 codes. 8 waves x 128-code strips; each
// wave reuses one B fragment across 4 independent M-subtile accumulators
// (4x less codebook L2 traffic, 4 hazard-free WMMAs back-to-back).
// ---------------------------------------------------------------------------
__global__ __launch_bounds__(256) void rvq_stage(
    const float* __restrict__ cbk,            // f32 codebooks [Q][N][D]
    float* __restrict__ residual,             // [B*T][D], updated in place
    const unsigned short* __restrict__ cb16,  // bf16 codebooks [Q][N][D]
    const float* __restrict__ c2,             // [Q][N] squared norms
    float* __restrict__ codes_out,            // [B][Q][T] as float
    float* __restrict__ lossp,                // [Q] partial sums
    const int q)
{
    __shared__ alignas(16) unsigned short lds_a[MTILE][512];  // 64 KB A tile (bf16)
    __shared__ float bestv_s[MTILE][8];
    __shared__ int   bestn_s[MTILE][8];
    __shared__ int   idx_s[MTILE];
    __shared__ float red[256];

    const int tid  = threadIdx.x;
    const int lane = tid & 31;
    const int w    = tid >> 5;               // wave 0..7
    const int m0   = blockIdx.x * MTILE;     // tile never crosses a batch (T%64==0)
    const int b    = m0 >> 11;               // / T_
    const int t0   = m0 & (T_ - 1);

    float* resrow = residual + (size_t)m0 * D_;

    // Stage residual tile into LDS as bf16 (coalesced f32 read, RNE convert).
    for (int i = tid; i < MTILE * 512; i += 256)
        ((unsigned short*)lds_a)[i] = f32_to_bf16(resrow[i]);
    __syncthreads();

    // A-fragment addressing (16x32 bf16, wave32 layout):
    //   lane l, row = l%16, runs at k0+(l/16)*8 and k0+16+(l/16)*8 (8 halves ea)
    const int arow = lane & 15;
    const int akb  = (lane >> 4) * 8;
    // B-fragment (32x16 bf16): lane l holds column n = n0+(l%16),
    //   runs at k0+(l/16)*8 and k0+16+(l/16)*8 (8 halves ea) in cb row n
    const int bkb  = (lane >> 4) * 8;

    float bestv[4][8];
    int   bestn[4][8];
#pragma unroll
    for (int mt = 0; mt < 4; ++mt)
#pragma unroll
        for (int j = 0; j < 8; ++j) { bestv[mt][j] = 3.4e38f; bestn[mt][j] = 0; }

    for (int nt = 0; nt < 8; ++nt) {
        const int n0 = w * 128 + nt * 16;
        const int n  = n0 + (lane & 15);
        const unsigned short* gb = cb16 + ((size_t)q * N_ + n) * D_ + bkb;

        v8f acc[4];
#pragma unroll
        for (int mt = 0; mt < 4; ++mt) acc[mt] = (v8f){};

#pragma unroll 2
        for (int k0 = 0; k0 < 512; k0 += 32) {
            ABFrag bf;
            bf.h[0] = *(const u16x8*)(gb + k0);
            bf.h[1] = *(const u16x8*)(gb + k0 + 16);
#pragma unroll
            for (int mt = 0; mt < 4; ++mt) {
                ABFrag a;
                a.h[0] = *(const u16x8*)&lds_a[mt * 16 + arow][k0 + akb];
                a.h[1] = *(const u16x8*)&lds_a[mt * 16 + arow][k0 + 16 + akb];
                acc[mt] = __builtin_amdgcn_wmma_f32_16x16x32_bf16(
                    false, a.v, false, bf.v, (short)0, acc[mt], false, false);
            }
        }
        // d2 argmin-equivalent value: c2[n] - 2*score (r^2 term constant per row)
        const float c2v = c2[q * N_ + n];
#pragma unroll
        for (int mt = 0; mt < 4; ++mt)
#pragma unroll
            for (int j = 0; j < 8; ++j) {
                float val = c2v - 2.0f * acc[mt][j];
                if (val < bestv[mt][j]) { bestv[mt][j] = val; bestn[mt][j] = n; }
            }
    }

    // Reduce across the 16 lanes that share a row (C layout: VGPR j ->
    // row j in lanes 0-15, row j+8 in lanes 16-31), per M-subtile.
#pragma unroll
    for (int mt = 0; mt < 4; ++mt) {
#pragma unroll
        for (int j = 0; j < 8; ++j) {
            float v = bestv[mt][j]; int n = bestn[mt][j];
            for (int m = 1; m < 16; m <<= 1) {
                float ov = __shfl_xor(v, m, 32);
                int   on = __shfl_xor(n, m, 32);
                if (ov < v) { v = ov; n = on; }
            }
            bestv[mt][j] = v; bestn[mt][j] = n;
        }
        if (lane == 0)
#pragma unroll
            for (int j = 0; j < 8; ++j) {
                bestv_s[mt * 16 + j][w] = bestv[mt][j];
                bestn_s[mt * 16 + j][w] = bestn[mt][j];
            }
        if (lane == 16)
#pragma unroll
            for (int j = 0; j < 8; ++j) {
                bestv_s[mt * 16 + j + 8][w] = bestv[mt][j];
                bestn_s[mt * 16 + j + 8][w] = bestn[mt][j];
            }
    }
    __syncthreads();

    // Cross-wave reduction + code output (as float, concatenated f32 d_out).
    if (tid < MTILE) {
        float v = bestv_s[tid][0]; int n = bestn_s[tid][0];
        for (int wv = 1; wv < 8; ++wv)
            if (bestv_s[tid][wv] < v) { v = bestv_s[tid][wv]; n = bestn_s[tid][wv]; }
        idx_s[tid] = n;
        codes_out[((size_t)b * Q_ + q) * T_ + t0 + tid] = (float)n;
    }
    __syncthreads();

    // residual -= cb[idx]; commit loss contribution = sum(residual_new^2)
    float part = 0.f;
    for (int i = tid; i < MTILE * 512; i += 256) {
        int r = i >> 9, d = i & 511;
        float nv = resrow[i] - cbk[((size_t)q * N_ + idx_s[r]) * D_ + d];
        resrow[i] = nv;
        part += nv * nv;
    }
    red[tid] = part;
    __syncthreads();
    for (int s = 128; s > 0; s >>= 1) {
        if (tid < s) red[tid] += red[tid + s];
        __syncthreads();
    }
    if (tid == 0) atomicAdd(&lossp[q], red[0]);
}

// ---------------------------------------------------------------------------
// Finalize: out[b][d][t] = z[b][d][t] - residual_final[b][t][d]
// (out == quantized transposed; straight-through cancels numerically)
// ---------------------------------------------------------------------------
__global__ __launch_bounds__(256) void rvq_finalize_out(
    const float* __restrict__ z, const float* __restrict__ residual,
    float* __restrict__ out)
{
    __shared__ float s[32][33];
    const int b  = blockIdx.z;
    const int t0 = blockIdx.x * 32;
    const int d0 = blockIdx.y * 32;
    const int tx = threadIdx.x & 31;
    const int ty = threadIdx.x >> 5;
#pragma unroll
    for (int i = 0; i < 4; ++i) {
        int tl = ty + 8 * i;
        s[tl][tx] = residual[((size_t)b * T_ + (t0 + tl)) * D_ + d0 + tx];  // s[t_local][d_local]
    }
    __syncthreads();
#pragma unroll
    for (int i = 0; i < 4; ++i) {
        int dl = ty + 8 * i;
        size_t off = ((size_t)b * D_ + (d0 + dl)) * T_ + t0 + tx;
        out[off] = z[off] - s[tx][dl];
    }
}

__global__ void rvq_loss(const float* __restrict__ lossp, float* __restrict__ out_loss)
{
    if (threadIdx.x == 0 && blockIdx.x == 0) {
        const float inv = 1.0f / (float)((size_t)B_ * T_ * D_);
        float total = 0.f;
        for (int q = 0; q < Q_; ++q) {
            float c = fminf(fmaxf(lossp[q] * inv, 0.f), 10.f);
            total += 0.25f * c;
        }
        out_loss[0] = fminf(fmaxf(total, 0.f), 10.f);
    }
}

// ---------------------------------------------------------------------------
extern "C" void kernel_launch(void* const* d_in, const int* in_sizes, int n_in,
                              void* d_out, int out_size, void* d_ws, size_t ws_size,
                              hipStream_t stream) {
    (void)in_sizes; (void)n_in; (void)out_size; (void)ws_size;
    const float* z   = (const float*)d_in[0];  // [B, D, T]
    const float* cbk = (const float*)d_in[1];  // [Q, N, D]

    float* out      = (float*)d_out;                        // [B, D, T]
    float* codes    = out + (size_t)B_ * D_ * T_;           // [B, Q, T] (float-encoded ints)
    float* out_loss = codes + (size_t)B_ * Q_ * T_;         // scalar

    char* ws = (char*)d_ws;
    float*          residual = (float*)ws;                                   // 64 MB
    unsigned short* cb16     = (unsigned short*)(ws + (size_t)67108864);     //  8 MB
    float*          c2       = (float*)(ws + (size_t)67108864 + 8388608);    // 32 KB
    float*          lossp    = (float*)(ws + (size_t)67108864 + 8388608 + 32768);

    dim3 tgrid(T_ / 32, D_ / 32, B_);
    rvq_transpose_in<<<tgrid, 256, 0, stream>>>(z, residual);
    rvq_prep_cb<<<(Q_ * N_) / 8, 256, 0, stream>>>(cbk, cb16, c2, lossp);
    for (int q = 0; q < Q_; ++q)
        rvq_stage<<<M_ / MTILE, 256, 0, stream>>>(cbk, residual, cb16, c2, codes, lossp, q);
    rvq_finalize_out<<<tgrid, 256, 0, stream>>>(z, residual, out);
    rvq_loss<<<1, 32, 0, stream>>>(lossp, out_loss);
}
